// CrossTransformerBlock_85023172592023
// MI455X (gfx1250) — matrix operand order repaired and checked by reading, not compiled, not run on hardware
//
#include <hip/hip_runtime.h>
#include <hip/hip_bf16.h>
#include <math.h>

typedef __attribute__((ext_vector_type(16))) __bf16 bf16x16;
typedef __attribute__((ext_vector_type(8)))  float  f32x8;

union FragU { uint4 q[2]; bf16x16 v; };

#if __has_builtin(__builtin_amdgcn_global_load_async_to_lds_b128) && \
    __has_builtin(__builtin_amdgcn_s_wait_asynccnt)
#define USE_ASYNC_LDS 1
#else
#define USE_ASYNC_LDS 0
#endif

#if USE_ASYNC_LDS
// builtin signature uses GCC-style int 4-vectors: (v4i AS1*, v4i AS3*, imm, imm)
typedef int v4i_g __attribute__((vector_size(16)));
typedef __attribute__((address_space(1))) v4i_g* g_v4i_ptr;
typedef __attribute__((address_space(3))) v4i_g* l_v4i_ptr;
// per-lane async copy of 16B global -> LDS, tracked by ASYNCcnt
static __device__ __forceinline__ void async_cp16(const void* g, void* l) {
  __builtin_amdgcn_global_load_async_to_lds_b128(
      (g_v4i_ptr)(unsigned long long)g,
      (l_v4i_ptr)(unsigned)(unsigned long long)l,
      0, 0);
}
#endif

static __device__ __forceinline__ unsigned short f2bf(float f) {
  unsigned int u = __float_as_uint(f);
  u += 0x7FFFu + ((u >> 16) & 1u);
  return (unsigned short)(u >> 16);
}

// ---------------- f32 -> bf16 convert ----------------
__global__ void cvt_bf16_kernel(const float* __restrict__ s,
                                unsigned short* __restrict__ d, int n) {
  int i = blockIdx.x * blockDim.x + threadIdx.x;
  if (i < n) d[i] = f2bf(s[i]);
}

// ---------------- LayerNorm over D=512, f32 in -> bf16 out ----------------
__global__ __launch_bounds__(256) void ln_bf16_kernel(
    const float* __restrict__ x, const float* __restrict__ g,
    const float* __restrict__ b, unsigned short* __restrict__ y) {
  const int D = 512;
  int row  = blockIdx.x * 8 + (threadIdx.x >> 5);
  int lane = threadIdx.x & 31;
  const float* xr = x + (size_t)row * D + lane * 16;
  float v[16];
  float s = 0.f, s2 = 0.f;
#pragma unroll
  for (int i = 0; i < 16; ++i) { v[i] = xr[i]; s += v[i]; s2 += v[i] * v[i]; }
#pragma unroll
  for (int m = 16; m >= 1; m >>= 1) {
    s  += __shfl_xor(s, m, 32);
    s2 += __shfl_xor(s2, m, 32);
  }
  float mean = s * (1.f / 512.f);
  float var  = s2 * (1.f / 512.f) - mean * mean;
  float rstd = rsqrtf(var + 1e-5f);
  int col = lane * 16;
  unsigned short* yr = y + (size_t)row * D + col;
#pragma unroll
  for (int i = 0; i < 16; ++i)
    yr[i] = f2bf((v[i] - mean) * rstd * g[col + i] + b[col + i]);
}

// ---------------- WMMA GEMM: C[M][N] = A[M][K] @ W[N][K]^T + bias ----------------
// A, W bf16 row-major. epi: 0 = f32 out, 1 = bf16 out, 2 = bf16 out + exact GELU
// 256 threads (8 waves, 4x2), tile 128x128, K-step 32, double-buffered async LDS
__global__ __launch_bounds__(256) void gemm_bf16_kernel(
    const unsigned short* __restrict__ A, const unsigned short* __restrict__ W,
    const float* __restrict__ bias, void* __restrict__ Cout,
    int M, int N, int K, int epi) {
#if USE_ASYNC_LDS
  __shared__ __align__(16) unsigned short As[2][128 * 32];
  __shared__ __align__(16) unsigned short Bs[2][128 * 32];
#else
  __shared__ __align__(16) unsigned short As[1][128 * 32];
  __shared__ __align__(16) unsigned short Bs[1][128 * 32];
#endif
  int tid  = threadIdx.x;
  int lane = tid & 31;
  int wave = tid >> 5;
  int wm = wave >> 1;   // 0..3 -> 32 rows each
  int wn = wave & 1;    // 0..1 -> 64 cols each
  int mBase = blockIdx.y * 128;
  int nBase = blockIdx.x * 128;

  f32x8 acc[2][4] = {};

  int lrow = tid >> 1;
  int lcol = (tid & 1) * 16;
  const unsigned short* Ag = A + (size_t)(mBase + lrow) * K + lcol;
  const unsigned short* Wg = W + (size_t)(nBase + lrow) * K + lcol;

  int rA = lane & 15;
  int hA = (lane >> 4) * 8;    // A frag K chunk offset
  int hB = (lane >> 4) * 16;   // unused (B frags are read with hA)

#if USE_ASYNC_LDS
  const int nIter = K >> 5;
  // issue one K-slab (4 x 16B per thread) into LDS buffer `buf`
  auto issue = [&](int buf, int k0) {
    async_cp16(Ag + k0,     &As[buf][lrow * 32 + lcol]);
    async_cp16(Ag + k0 + 8, &As[buf][lrow * 32 + lcol + 8]);
    async_cp16(Wg + k0,     &Bs[buf][lrow * 32 + lcol]);
    async_cp16(Wg + k0 + 8, &Bs[buf][lrow * 32 + lcol + 8]);
  };
  issue(0, 0);
  if (nIter > 1) issue(1, 32);
  for (int i = 0; i < nIter; ++i) {
    if (i + 1 < nIter) __builtin_amdgcn_s_wait_asynccnt(4);  // slab i done, keep i+1 in flight
    else               __builtin_amdgcn_s_wait_asynccnt(0);
    __syncthreads();
    int buf = i & 1;
    FragU fa[2], fb[4];
#pragma unroll
    for (int mt = 0; mt < 2; ++mt) {
      const unsigned short* p = &As[buf][(wm * 32 + mt * 16 + rA) * 32];
      fa[mt].q[0] = *(const uint4*)(p + hA);
      fa[mt].q[1] = *(const uint4*)(p + 16 + hA);
    }
#pragma unroll
    for (int nt = 0; nt < 4; ++nt) {
      const unsigned short* p = &Bs[buf][(wn * 64 + nt * 16 + rA) * 32];
      fb[nt].q[0] = *(const uint4*)(p + hA);
      fb[nt].q[1] = *(const uint4*)(p + 16 + hA);
    }
#pragma unroll
    for (int mt = 0; mt < 2; ++mt)
#pragma unroll
      for (int nt = 0; nt < 4; ++nt)
        acc[mt][nt] = __builtin_amdgcn_wmma_f32_16x16x32_bf16(
            false, fa[mt].v, false, fb[nt].v, (short)0, acc[mt][nt], false, false);
    __syncthreads();  // all waves done reading `buf` before it is refilled
    if (i + 2 < nIter) issue(buf, (i + 2) * 32);
  }
#else
  for (int k0 = 0; k0 < K; k0 += 32) {
    uint4 a0 = *(const uint4*)(Ag + k0);
    uint4 a1 = *(const uint4*)(Ag + k0 + 8);
    uint4 b0 = *(const uint4*)(Wg + k0);
    uint4 b1 = *(const uint4*)(Wg + k0 + 8);
    if (k0 + 32 < K) {
      __builtin_prefetch(Ag + k0 + 32, 0, 1);
      __builtin_prefetch(Wg + k0 + 32, 0, 1);
    }
    __syncthreads();
    *(uint4*)&As[0][lrow * 32 + lcol]     = a0;
    *(uint4*)&As[0][lrow * 32 + lcol + 8] = a1;
    *(uint4*)&Bs[0][lrow * 32 + lcol]     = b0;
    *(uint4*)&Bs[0][lrow * 32 + lcol + 8] = b1;
    __syncthreads();

    FragU fa[2], fb[4];
#pragma unroll
    for (int mt = 0; mt < 2; ++mt) {
      const unsigned short* p = &As[0][(wm * 32 + mt * 16 + rA) * 32];
      fa[mt].q[0] = *(const uint4*)(p + hA);
      fa[mt].q[1] = *(const uint4*)(p + 16 + hA);
    }
#pragma unroll
    for (int nt = 0; nt < 4; ++nt) {
      const unsigned short* p = &Bs[0][(wn * 64 + nt * 16 + rA) * 32];
      fb[nt].q[0] = *(const uint4*)(p + hA);
      fb[nt].q[1] = *(const uint4*)(p + 16 + hA);
    }
#pragma unroll
    for (int mt = 0; mt < 2; ++mt)
#pragma unroll
      for (int nt = 0; nt < 4; ++nt)
        acc[mt][nt] = __builtin_amdgcn_wmma_f32_16x16x32_bf16(
            false, fa[mt].v, false, fb[nt].v, (short)0, acc[mt][nt], false, false);
  }
#endif

#pragma unroll
  for (int mt = 0; mt < 2; ++mt)
#pragma unroll
    for (int nt = 0; nt < 4; ++nt) {
      int n = nBase + wn * 64 + nt * 16 + rA;
      float bv = bias[n];
      int m0 = mBase + wm * 32 + mt * 16 + (lane >> 4) * 8;
#pragma unroll
      for (int v = 0; v < 8; ++v) {
        float x = acc[mt][nt][v] + bv;
        if (epi == 2) x = 0.5f * x * (1.f + erff(x * 0.70710678118654752f));
        size_t idx = (size_t)(m0 + v) * N + n;
        if (epi == 0) ((float*)Cout)[idx] = x;
        else          ((unsigned short*)Cout)[idx] = f2bf(x);
      }
    }
}

// ---------------- Flash attention (bf16 QKV in, bf16 out), per (b,h,128-q tile) ----
__global__ __launch_bounds__(128) void attn_kernel(
    const unsigned short* __restrict__ Q, const unsigned short* __restrict__ Kt,
    const unsigned short* __restrict__ V, unsigned short* __restrict__ O) {
  const int Nq = 1024, D = 512;
  const float scale = 0.125f;   // 1/sqrt(64)
  __shared__ __align__(16) unsigned short Qs[128 * 64];
  __shared__ __align__(16) unsigned short Ks[64 * 64];
  __shared__ __align__(16) unsigned short Vt[64 * 64];     // transposed: [dk][key]
  __shared__ __align__(16) unsigned short Ps[4][32 * 64];  // per-wave P scratch

  int tid  = threadIdx.x;
  int lane = tid & 31;
  int wave = tid >> 5;
  int b = blockIdx.z, h = blockIdx.y;
  int qBase = blockIdx.x * 128;
  size_t base = ((size_t)b * Nq) * D + h * 64;

  {  // load 128x64 Q tile
    const unsigned short* src = Q + base + (size_t)(qBase + tid) * D;
    unsigned short* dst = &Qs[tid * 64];
#pragma unroll
    for (int j = 0; j < 8; ++j)
#if USE_ASYNC_LDS
      async_cp16(src + j * 8, dst + j * 8);
#else
      *(uint4*)(dst + j * 8) = *(const uint4*)(src + j * 8);
#endif
  }

  f32x8 o[2][4] = {};
  float mrow[2][8], lrow[2][8];
#pragma unroll
  for (int mt = 0; mt < 2; ++mt)
#pragma unroll
    for (int v = 0; v < 8; ++v) { mrow[mt][v] = -1e30f; lrow[mt][v] = 0.f; }

  int rA  = lane & 15;
  int hAo = (lane >> 4) * 8;
  int hBo = (lane >> 4) * 16;

  for (int kt = 0; kt < Nq; kt += 64) {
    {  // load 64x64 K tile (row-major, async) and V tile transposed (VALU path)
      int row = tid >> 1, c = (tid & 1) * 32;
      const unsigned short* ksrc = Kt + base + (size_t)(kt + row) * D + c;
      unsigned short* kdst = &Ks[row * 64 + c];
#pragma unroll
      for (int j = 0; j < 4; ++j)
#if USE_ASYNC_LDS
        async_cp16(ksrc + j * 8, kdst + j * 8);
#else
        *(uint4*)(kdst + j * 8) = *(const uint4*)(ksrc + j * 8);
#endif
      const unsigned short* vsrc = V + base + (size_t)(kt + row) * D + c;
      uint4 t[4];
#pragma unroll
      for (int j = 0; j < 4; ++j) t[j] = *(const uint4*)(vsrc + j * 8);
      const unsigned short* ts = (const unsigned short*)t;
#pragma unroll
      for (int j = 0; j < 32; ++j)
        Vt[(c + j) * 64 + row] = ts[j];
    }
#if USE_ASYNC_LDS
    __builtin_amdgcn_s_wait_asynccnt(0);   // Q (first iter) + K tile landed in LDS
#endif
    __syncthreads();

    // S = Q @ K^T : 32 queries x 64 keys per wave
    f32x8 s[2][4] = {};
#pragma unroll
    for (int kk = 0; kk < 64; kk += 32) {
      FragU fa[2], fb[4];
#pragma unroll
      for (int mt = 0; mt < 2; ++mt) {
        const unsigned short* p = &Qs[(wave * 32 + mt * 16 + rA) * 64 + kk];
        fa[mt].q[0] = *(const uint4*)(p + hAo);
        fa[mt].q[1] = *(const uint4*)(p + 16 + hAo);
      }
#pragma unroll
      for (int nt = 0; nt < 4; ++nt) {
        const unsigned short* p = &Ks[(nt * 16 + rA) * 64 + kk];
        fb[nt].q[0] = *(const uint4*)(p + hAo);
        fb[nt].q[1] = *(const uint4*)(p + 16 + hAo);
      }
#pragma unroll
      for (int mt = 0; mt < 2; ++mt)
#pragma unroll
        for (int nt = 0; nt < 4; ++nt)
          s[mt][nt] = __builtin_amdgcn_wmma_f32_16x16x32_bf16(
              false, fa[mt].v, false, fb[nt].v, (short)0, s[mt][nt], false, false);
    }

    // online softmax (row = mt*16 + (lane>>4)*8 + v; reduce inside 16-lane halves)
#pragma unroll
    for (int mt = 0; mt < 2; ++mt)
#pragma unroll
      for (int v = 0; v < 8; ++v) {
        float smax = -1e30f;
#pragma unroll
        for (int nt = 0; nt < 4; ++nt) smax = fmaxf(smax, s[mt][nt][v] * scale);
#pragma unroll
        for (int msk = 8; msk >= 1; msk >>= 1)
          smax = fmaxf(smax, __shfl_xor(smax, msk, 32));
        float mnew  = fmaxf(mrow[mt][v], smax);
        float alpha = __expf(mrow[mt][v] - mnew);
        mrow[mt][v] = mnew;
        float rs = 0.f;
#pragma unroll
        for (int nt = 0; nt < 4; ++nt) {
          float p = __expf(s[mt][nt][v] * scale - mnew);
          s[mt][nt][v] = p;
          rs += p;
        }
#pragma unroll
        for (int msk = 8; msk >= 1; msk >>= 1) rs += __shfl_xor(rs, msk, 32);
        lrow[mt][v] = lrow[mt][v] * alpha + rs;
#pragma unroll
        for (int nt = 0; nt < 4; ++nt) o[mt][nt][v] *= alpha;
      }

    // stage P (bf16) through per-wave LDS to re-layout C -> A fragment
    unsigned short* Pw = Ps[wave];
#pragma unroll
    for (int mt = 0; mt < 2; ++mt)
#pragma unroll
      for (int nt = 0; nt < 4; ++nt) {
        int col = nt * 16 + rA;
        int r0  = mt * 16 + (lane >> 4) * 8;
#pragma unroll
        for (int v = 0; v < 8; ++v)
          Pw[(r0 + v) * 64 + col] = f2bf(s[mt][nt][v]);
      }

    // O += P @ V
#pragma unroll
    for (int kk = 0; kk < 64; kk += 32) {
      FragU fa[2], fb[4];
#pragma unroll
      for (int mt = 0; mt < 2; ++mt) {
        const unsigned short* p = &Pw[(mt * 16 + rA) * 64 + kk];
        fa[mt].q[0] = *(const uint4*)(p + hAo);
        fa[mt].q[1] = *(const uint4*)(p + 16 + hAo);
      }
#pragma unroll
      for (int nt = 0; nt < 4; ++nt) {
        const unsigned short* p = &Vt[(nt * 16 + rA) * 64 + kk];
        fb[nt].q[0] = *(const uint4*)(p + hAo);
        fb[nt].q[1] = *(const uint4*)(p + 16 + hAo);
      }
#pragma unroll
      for (int mt = 0; mt < 2; ++mt)
#pragma unroll
        for (int nt = 0; nt < 4; ++nt)
          o[mt][nt] = __builtin_amdgcn_wmma_f32_16x16x32_bf16(
              false, fa[mt].v, false, fb[nt].v, (short)0, o[mt][nt], false, false);
    }
    __syncthreads();
  }

  // normalize and store bf16
#pragma unroll
  for (int mt = 0; mt < 2; ++mt)
#pragma unroll
    for (int nt = 0; nt < 4; ++nt) {
      int col = nt * 16 + rA;
      int r0  = qBase + wave * 32 + mt * 16 + (lane >> 4) * 8;
#pragma unroll
      for (int v = 0; v < 8; ++v) {
        float val = o[mt][nt][v] / lrow[mt][v];
        O[base + (size_t)(r0 + v) * D + col] = f2bf(val);
      }
    }
}

// ---------------- elementwise ----------------
__global__ void combine_x_kernel(const float* __restrict__ base, const float* __restrict__ p,
                                 const float* __restrict__ q, const float* __restrict__ coef,
                                 int ia, int ib, float* __restrict__ out, int n) {
  int i = blockIdx.x * blockDim.x + threadIdx.x;
  if (i < n) out[i] = coef[ia] * base[i] + 0.5f * coef[ib] * (p[i] + q[i]);
}

__global__ void final_out_kernel(const float* __restrict__ x, const float* __restrict__ m,
                                 const float* __restrict__ coef, int ia, int ib,
                                 float* __restrict__ out, int n) {
  int i = blockIdx.x * blockDim.x + threadIdx.x;
  if (i < n) out[i] = coef[ia] * x[i] + coef[ib] * m[i];
}

extern "C" void kernel_launch(void* const* d_in, const int* in_sizes, int n_in,
                              void* d_out, int out_size, void* d_ws, size_t ws_size,
                              hipStream_t stream) {
  const int BN = 4096, D = 512, E = 2048;
  const size_t MB = 1ull << 20;

  const float* rgb = (const float*)d_in[0];
  const float* ir  = (const float*)d_in[1];
  const float* nir = (const float*)d_in[2];
  const float* aln1_g = (const float*)d_in[4];
  const float* aln1_b = (const float*)d_in[5];
  const float* aln2_g = (const float*)d_in[6];
  const float* aln2_b = (const float*)d_in[7];
  const float* bq  = (const float*)d_in[9];
  const float* bk  = (const float*)d_in[11];
  const float* bv  = (const float*)d_in[13];
  const float* bo  = (const float*)d_in[15];
  const float* boi = (const float*)d_in[17];
  const float* ln1_g = (const float*)d_in[18];
  const float* ln1_b = (const float*)d_in[19];
  const float* ln2_g = (const float*)d_in[20];
  const float* ln2_b = (const float*)d_in[21];
  const float* ln3_g = (const float*)d_in[22];
  const float* ln3_b = (const float*)d_in[23];
  const float* vb1 = (const float*)d_in[25];
  const float* vb2 = (const float*)d_in[27];
  const float* ib1 = (const float*)d_in[29];
  const float* ib2 = (const float*)d_in[31];
  const float* nb1 = (const float*)d_in[33];
  const float* nb2 = (const float*)d_in[35];
  const float* coef = (const float*)d_in[36];

  char* ws = (char*)d_ws;

  // bf16 weights (persistent, region [0, 16MB))
  unsigned short* wq_b  = (unsigned short*)ws;
  unsigned short* wk_b  = wq_b  + 512 * 512;
  unsigned short* wv_b  = wk_b  + 512 * 512;
  unsigned short* wo_b  = wv_b  + 512 * 512;
  unsigned short* woi_b = wo_b  + 512 * 512;
  unsigned short* vW1_b = woi_b + 512 * 512;
  unsigned short* iW1_b = vW1_b + 2048 * 512;
  unsigned short* nW1_b = iW1_b + 2048 * 512;
  unsigned short* vW2_b = nW1_b + 2048 * 512;
  unsigned short* iW2_b = vW2_b + 512 * 2048;
  unsigned short* nW2_b = iW2_b + 512 * 2048;

  // LN'd activations bf16 [16MB, 32MB)
  unsigned short* ln_rgb1 = (unsigned short*)(ws + 16 * MB);
  unsigned short* ln_ir1  = ln_rgb1 + BN * D;
  unsigned short* ln_ir2  = ln_ir1  + BN * D;
  unsigned short* ln_nir2 = ln_ir2  + BN * D;

  // Q/K/V bf16 [32MB, 80MB)
  unsigned short* qkv = (unsigned short*)(ws + 32 * MB);
  unsigned short* q_rgb1 = qkv + 0  * BN * D;
  unsigned short* k_rgb1 = qkv + 1  * BN * D;
  unsigned short* v_rgb1 = qkv + 2  * BN * D;
  unsigned short* q_ir1  = qkv + 3  * BN * D;
  unsigned short* k_ir1  = qkv + 4  * BN * D;
  unsigned short* v_ir1  = qkv + 5  * BN * D;
  unsigned short* q_ir2  = qkv + 6  * BN * D;
  unsigned short* k_ir2  = qkv + 7  * BN * D;
  unsigned short* v_ir2  = qkv + 8  * BN * D;
  unsigned short* q_nir2 = qkv + 9  * BN * D;
  unsigned short* k_nir2 = qkv + 10 * BN * D;
  unsigned short* v_nir2 = qkv + 11 * BN * D;

  // attention outputs bf16 [80MB, 104MB)
  unsigned short* ao = (unsigned short*)(ws + 80 * MB);
  unsigned short* a_ra = ao + 0 * BN * D;
  unsigned short* a_ta = ao + 1 * BN * D;
  unsigned short* a_rb = ao + 2 * BN * D;
  unsigned short* a_na = ao + 3 * BN * D;
  unsigned short* a_tb = ao + 4 * BN * D;
  unsigned short* a_nb = ao + 5 * BN * D;

  // output-projection f32 [32MB, 80MB) (reuses QKV region, free after attention)
  float* ra_f = (float*)(ws + 32 * MB);
  float* ta_f = ra_f + BN * D;
  float* rb_f = ta_f + BN * D;
  float* na_f = rb_f + BN * D;
  float* tb_f = na_f + BN * D;
  float* nb_f = tb_f + BN * D;

  // x f32 [104MB, 128MB)
  float* xr = (float*)(ws + 104 * MB);
  float* xt = xr + BN * D;
  float* xn = xt + BN * D;

  // ln(x) bf16 (reuses LN region [16MB, 28MB))
  unsigned short* lnr = (unsigned short*)(ws + 16 * MB);
  unsigned short* lnt = lnr + BN * D;
  unsigned short* lnn = lnt + BN * D;

  // MLP hidden bf16 (reuses [32MB, 80MB), free after combine)
  unsigned short* hr = (unsigned short*)(ws + 32 * MB);
  unsigned short* ht = hr + BN * E;
  unsigned short* hn = ht + BN * E;

  // MLP out f32 (reuses [80MB, 104MB))
  float* mr = (float*)(ws + 80 * MB);
  float* mt = mr + BN * D;
  float* mn = mt + BN * D;

  float* out_r = (float*)d_out;
  float* out_t = out_r + BN * D;
  float* out_n = out_t + BN * D;

  auto cvt = [&](const void* s, unsigned short* dd, int n) {
    cvt_bf16_kernel<<<(n + 255) / 256, 256, 0, stream>>>((const float*)s, dd, n);
  };
  auto ln = [&](const float* x, const float* g, const float* b, unsigned short* y) {
    ln_bf16_kernel<<<BN / 8, 256, 0, stream>>>(x, g, b, y);
  };
  auto gemm = [&](const unsigned short* A, const unsigned short* W, const float* bias,
                  void* C, int M, int N, int K, int epi) {
    dim3 grid(N / 128, M / 128);
    gemm_bf16_kernel<<<grid, 256, 0, stream>>>(A, W, bias, C, M, N, K, epi);
  };
  auto attn = [&](const unsigned short* q, const unsigned short* k,
                  const unsigned short* v, unsigned short* o) {
    dim3 grid(8, 8, 4);  // (qtile, H, B)
    attn_kernel<<<grid, 128, 0, stream>>>(q, k, v, o);
  };

  // 1) convert weights to bf16
  cvt(d_in[8],  wq_b,  512 * 512);
  cvt(d_in[10], wk_b,  512 * 512);
  cvt(d_in[12], wv_b,  512 * 512);
  cvt(d_in[14], wo_b,  512 * 512);
  cvt(d_in[16], woi_b, 512 * 512);
  cvt(d_in[24], vW1_b, 2048 * 512);
  cvt(d_in[28], iW1_b, 2048 * 512);
  cvt(d_in[32], nW1_b, 2048 * 512);
  cvt(d_in[26], vW2_b, 512 * 2048);
  cvt(d_in[30], iW2_b, 512 * 2048);
  cvt(d_in[34], nW2_b, 512 * 2048);

  // 2) layernorms (aln1 on 'a' role, aln2 on 'b' role)
  ln(rgb, aln1_g, aln1_b, ln_rgb1);
  ln(ir,  aln1_g, aln1_b, ln_ir1);
  ln(ir,  aln2_g, aln2_b, ln_ir2);
  ln(nir, aln2_g, aln2_b, ln_nir2);

  // 3) QKV projections (bf16 out)
  const unsigned short* lns[4] = {ln_rgb1, ln_ir1, ln_ir2, ln_nir2};
  unsigned short* qs[4] = {q_rgb1, q_ir1, q_ir2, q_nir2};
  unsigned short* ks[4] = {k_rgb1, k_ir1, k_ir2, k_nir2};
  unsigned short* vs[4] = {v_rgb1, v_ir1, v_ir2, v_nir2};
  for (int t = 0; t < 4; ++t) {
    gemm(lns[t], wq_b, bq, qs[t], BN, D, D, 1);
    gemm(lns[t], wk_b, bk, ks[t], BN, D, D, 1);
    gemm(lns[t], wv_b, bv, vs[t], BN, D, D, 1);
  }

  // 4) six attentions
  attn(q_ir2,  k_rgb1, v_rgb1, a_ra);  // cross(rgb,ir)  att_a
  attn(q_rgb1, k_ir2,  v_ir2,  a_ta);  // cross(rgb,ir)  att_b
  attn(q_nir2, k_rgb1, v_rgb1, a_rb);  // cross(rgb,nir) att_a
  attn(q_rgb1, k_nir2, v_nir2, a_na);  // cross(rgb,nir) att_b
  attn(q_nir2, k_ir1,  v_ir1,  a_tb);  // cross(ir,nir)  att_a
  attn(q_ir1,  k_nir2, v_nir2, a_nb);  // cross(ir,nir)  att_b

  // 5) output projections (f32 out)
  gemm(a_ra, wo_b,  bo,  ra_f, BN, D, D, 0);
  gemm(a_ta, woi_b, boi, ta_f, BN, D, D, 0);
  gemm(a_rb, wo_b,  bo,  rb_f, BN, D, D, 0);
  gemm(a_na, woi_b, boi, na_f, BN, D, D, 0);
  gemm(a_tb, wo_b,  bo,  tb_f, BN, D, D, 0);
  gemm(a_nb, woi_b, boi, nb_f, BN, D, D, 0);

  // 6) x = coef_a * base + coef_b * 0.5 * (p + q)
  int nelem = BN * D;
  int eg = (nelem + 255) / 256;
  combine_x_kernel<<<eg, 256, 0, stream>>>(rgb, ra_f, rb_f, coef, 0, 1, xr, nelem);
  combine_x_kernel<<<eg, 256, 0, stream>>>(ir,  ta_f, tb_f, coef, 2, 3, xt, nelem);
  combine_x_kernel<<<eg, 256, 0, stream>>>(nir, na_f, nb_f, coef, 8, 9, xn, nelem);

  // 7) MLPs: ln -> W1^T + b1 (GELU, bf16) -> W2^T + b2 (f32)
  ln(xr, ln1_g, ln1_b, lnr);
  ln(xt, ln2_g, ln2_b, lnt);
  ln(xn, ln3_g, ln3_b, lnn);
  gemm(lnr, vW1_b, vb1, hr, BN, E, D, 2);
  gemm(lnt, iW1_b, ib1, ht, BN, E, D, 2);
  gemm(lnn, nW1_b, nb1, hn, BN, E, D, 2);
  gemm(hr, vW2_b, vb2, mr, BN, D, E, 0);
  gemm(ht, iW2_b, ib2, mt, BN, D, E, 0);
  gemm(hn, nW2_b, nb2, mn, BN, D, E, 0);

  // 8) final residual: out = coef_a * x + coef_b * mlp
  final_out_kernel<<<eg, 256, 0, stream>>>(xr, mr, coef, 4,  5,  out_r, nelem);
  final_out_kernel<<<eg, 256, 0, stream>>>(xt, mt, coef, 6,  7,  out_t, nelem);
  final_out_kernel<<<eg, 256, 0, stream>>>(xn, mn, coef, 10, 11, out_n, nelem);
}
